// GNNMolEncoder_80290118631455
// MI455X (gfx1250) — compile-verified, matrix operand order checked by reading, not run on hardware
//
#include <hip/hip_runtime.h>
#include <cstddef>
#include <cstdint>

// ---------------------------------------------------------------------------
// Types
// ---------------------------------------------------------------------------
typedef __bf16 v16bf __attribute__((ext_vector_type(16)));
typedef __bf16 v4bf  __attribute__((ext_vector_type(4)));
typedef float  v8f   __attribute__((ext_vector_type(8)));
typedef float  f32x4 __attribute__((ext_vector_type(4)));
typedef unsigned int u32x4 __attribute__((ext_vector_type(4)));

union FragU { u32x4 q[2]; v16bf v; };

// ---------------------------------------------------------------------------
// Model constants
// ---------------------------------------------------------------------------
constexpr int Hh  = 348;
constexpr int NHh = 12;
constexpr int HDh = 29;
constexpr int FFh = 696;
constexpr int Nn  = 16384;
constexpr int En  = 32768;
constexpr int Gn  = 512;
constexpr float AVG_LOG = 1.2182999f;   // sum(log(d+1)*hist)/15 for hist [1..5]

// GEMM tiling: block 64x128, K-step 64 (2 WMMA-K chunks), 8 waves of 32x32.
constexpr int BM = 64, BN = 128, BK = 64, LDK = 72;  // LDK*2B = 144B pitch (16B-aligned, conflict-free)

// ---------------------------------------------------------------------------
// Device helpers
// ---------------------------------------------------------------------------
__device__ __forceinline__ float gelu_f(float x) {
    return 0.5f * x * (1.0f + erff(x * 0.70710678118f));
}

__device__ __forceinline__ void atomMaxF(float* a, float v) {
    int old = __float_as_int(*a);
    while (__int_as_float(old) < v) {
        int prev = atomicCAS((int*)a, old, __float_as_int(v));
        if (prev == old) break;
        old = prev;
    }
}
__device__ __forceinline__ void atomMinF(float* a, float v) {
    int old = __float_as_int(*a);
    while (__int_as_float(old) > v) {
        int prev = atomicCAS((int*)a, old, __float_as_int(v));
        if (prev == old) break;
        old = prev;
    }
}

// ---------------------------------------------------------------------------
// Weight prep: W[K,N] f32  ->  Wt[Nt,Kp] bf16 (transposed, zero padded)
// ---------------------------------------------------------------------------
__global__ void __launch_bounds__(256)
k_prep_w(const float* __restrict__ W, __bf16* __restrict__ Wt,
         int K, int N, int Kp, int Nt) {
    size_t i = (size_t)blockIdx.x * 256 + threadIdx.x;
    size_t tot = (size_t)Nt * Kp;
    if (i >= tot) return;
    int n = (int)(i / Kp), k = (int)(i % Kp);
    float v = (n < N && k < K) ? W[(size_t)k * N + n] : 0.0f;
    Wt[i] = (__bf16)v;
}

// ---------------------------------------------------------------------------
// BF16 WMMA GEMM: C[M,N] = act( A[M,K] @ Wt^T + bias )
//   A:  f32 row-major (converted to bf16 while staging into LDS)
//   Wt: bf16 [Nt,Kp] (row n holds column n of W, K-contiguous, Kp%64==0)
// B tile staged with GLOBAL_LOAD_ASYNC_TO_LDS_B128 (ASYNCcnt), overlapped
// with the WMMA work of the current tile; drained with s_wait_asynccnt 0
// before the workgroup barrier.
// ---------------------------------------------------------------------------
__global__ void __launch_bounds__(256)
k_gemm(const float* __restrict__ A, const __bf16* __restrict__ Bt,
       const float* __restrict__ bias, float* __restrict__ C,
       int M, int N, int K, int Kp, int act) {
    __shared__ __bf16 sA[2][BM * LDK];
    __shared__ __bf16 sB[2][BN * LDK];

    const int tid  = threadIdx.x;
    const int bm   = blockIdx.y * BM;
    const int bn   = blockIdx.x * BN;
    const int lane = tid & 31;
    const int wid  = tid >> 5;
    const int mW   = (wid & 1) * 32;
    const int nW   = (wid >> 1) * 32;
    const int l15  = lane & 15;
    const int hsel = lane >> 4;

    v8f acc[2][2] = {};

    const int ksteps = (K + BK - 1) / BK;

    auto stageA = [&](int buf, int k0) {
#pragma unroll
        for (int it = 0; it < 4; ++it) {
            int idx = tid + it * 256;          // 1024 quad-float slots (64 rows x 16)
            int r   = idx >> 4;                // 0..63
            int c4  = (idx & 15) << 2;         // 0,4,...,60
            int gr = bm + r, gc = k0 + c4;
            f32x4 v = {0.f, 0.f, 0.f, 0.f};
            if (gr < M) {
                const float* p = A + (size_t)gr * K + gc;
                if (gc + 3 < K && ((((size_t)p) & 15) == 0)) {
                    v = *(const f32x4*)p;
                } else {
                    if (gc + 0 < K) v.x = p[0];
                    if (gc + 1 < K) v.y = p[1];
                    if (gc + 2 < K) v.z = p[2];
                    if (gc + 3 < K) v.w = p[3];
                }
            }
            v4bf o = {(__bf16)v.x, (__bf16)v.y, (__bf16)v.z, (__bf16)v.w};
            *(v4bf*)&sA[buf][r * LDK + c4] = o;
        }
    };
    // Async copy of the bf16 weight tile straight into LDS (no VGPR data path).
    auto stageB = [&](int buf, int k0) {
#pragma unroll
        for (int it = 0; it < 4; ++it) {
            int idx = tid + it * 256;          // 1024 slots of 16B (128 rows x 8)
            int r   = idx >> 3;                // 0..127
            int c8  = (idx & 7) << 3;          // 0,8,...,56
            const __bf16* gp = Bt + (size_t)(bn + r) * Kp + (k0 + c8);
            unsigned long long ga = (unsigned long long)(uintptr_t)gp;
            unsigned la = (unsigned)(uintptr_t)&sB[buf][r * LDK + c8];
            asm volatile("global_load_async_to_lds_b128 %0, %1, off"
                         :: "v"(la), "v"(ga) : "memory");
        }
    };
    auto waitAsync = [&]() {
        asm volatile("s_wait_asynccnt 0x0" ::: "memory");
    };
    auto mma = [&](int buf) {
#pragma unroll
        for (int kc = 0; kc < 2; ++kc) {       // two 16x16x32 K-chunks per stage
            v16bf af[2], bfr[2];
#pragma unroll
            for (int mf = 0; mf < 2; ++mf) {
                const __bf16* ap = &sA[buf][(mW + mf * 16 + l15) * LDK + kc * 32 + hsel * 8];
                FragU u; u.q[0] = *(const u32x4*)ap; u.q[1] = *(const u32x4*)(ap + 16);
                af[mf] = u.v;
            }
#pragma unroll
            for (int nf = 0; nf < 2; ++nf) {
                const __bf16* bp = &sB[buf][(nW + nf * 16 + l15) * LDK + kc * 32 + hsel * 8];
                FragU u; u.q[0] = *(const u32x4*)bp; u.q[1] = *(const u32x4*)(bp + 16);
                bfr[nf] = u.v;
            }
#pragma unroll
            for (int mf = 0; mf < 2; ++mf)
#pragma unroll
                for (int nf = 0; nf < 2; ++nf)
                    acc[mf][nf] = __builtin_amdgcn_wmma_f32_16x16x32_bf16(
                        false, af[mf], false, bfr[nf], (short)0, acc[mf][nf],
                        false, false);
        }
    };

    stageA(0, 0); stageB(0, 0);
    waitAsync();
    __syncthreads();
    for (int ks = 0; ks < ksteps; ++ks) {
        int cur = ks & 1, nxt = cur ^ 1;
        if (ks + 1 < ksteps) { stageA(nxt, (ks + 1) * BK); stageB(nxt, (ks + 1) * BK); }
        mma(cur);
        waitAsync();           // drain next-tile async copies (overlapped with mma)
        __syncthreads();
    }

    // epilogue: C/D layout — VGPR r: lanes 0-15 -> row r, lanes 16-31 -> row r+8
#pragma unroll
    for (int nf = 0; nf < 2; ++nf) {
        int col = bn + nW + nf * 16 + l15;
        if (col >= N) continue;
        float bv = bias ? bias[col] : 0.0f;
#pragma unroll
        for (int mf = 0; mf < 2; ++mf) {
            int rbase = bm + mW + mf * 16 + hsel * 8;
#pragma unroll
            for (int r = 0; r < 8; ++r) {
                int row = rbase + r;
                if (row < M) {
                    float vv = acc[mf][nf][r] + bv;
                    if (act == 1)      vv = gelu_f(vv);
                    else if (act == 2) vv = fmaxf(vv, 0.0f);
                    C[(size_t)row * N + col] = vv;
                }
            }
        }
    }
}

// ---------------------------------------------------------------------------
// LayerNorm (row-per-block): Y = post( LN(X)*g + b + add )
// ---------------------------------------------------------------------------
__global__ void __launch_bounds__(256)
k_ln(const float* __restrict__ X, const float* __restrict__ gw,
     const float* __restrict__ bw, const float* __restrict__ add,
     float* __restrict__ Y, int D, int out_ld, int out_off, int post) {
    __shared__ float red[256];
    const int tid = threadIdx.x;
    const float* x = X + (size_t)blockIdx.x * D;
    float s = 0.f;
    for (int i = tid; i < D; i += 256) s += x[i];
    red[tid] = s; __syncthreads();
    for (int o = 128; o > 0; o >>= 1) { if (tid < o) red[tid] += red[tid + o]; __syncthreads(); }
    float mean = red[0] / D;
    __syncthreads();
    float v = 0.f;
    for (int i = tid; i < D; i += 256) { float d = x[i] - mean; v += d * d; }
    red[tid] = v; __syncthreads();
    for (int o = 128; o > 0; o >>= 1) { if (tid < o) red[tid] += red[tid + o]; __syncthreads(); }
    float rstd = rsqrtf(red[0] / D + 1e-5f);
    float* y = Y + (size_t)blockIdx.x * out_ld + out_off;
    const float* ad = add ? add + (size_t)blockIdx.x * D : nullptr;
    for (int i = tid; i < D; i += 256) {
        float t = (x[i] - mean) * rstd * gw[i] + bw[i];
        if (ad) t += ad[i];
        if (post) t = gelu_f(t);
        y[i] = t;
    }
}

// ---------------------------------------------------------------------------
// Elementwise / scatter kernels
// ---------------------------------------------------------------------------
__global__ void __launch_bounds__(256)
k_fill(float* p, float v, size_t n) {
    size_t i = (size_t)blockIdx.x * 256 + threadIdx.x;
    if (i < n) p[i] = v;
}
__global__ void __launch_bounds__(256)
k_add(const float* a, const float* b, float* y, size_t n) {
    size_t i = (size_t)blockIdx.x * 256 + threadIdx.x;
    if (i < n) y[i] = a[i] + b[i];
}
__global__ void __launch_bounds__(256)
k_add_rowvec(float* Y, const float* bias, int R, int D) {
    size_t i = (size_t)blockIdx.x * 256 + threadIdx.x;
    if (i >= (size_t)R * D) return;
    Y[i] += bias[i % D];
}
__global__ void __launch_bounds__(256)
k_deg(const int* dst, float* deg, int E) {
    int e = blockIdx.x * 256 + threadIdx.x;
    if (e < E) atomicAdd(&deg[dst[e]], 1.0f);
}
__global__ void __launch_bounds__(256)
k_seg_sum(const float* in, const int* idx, float* out, int R, int D, int sq) {
    size_t i = (size_t)blockIdx.x * 256 + threadIdx.x;
    if (i >= (size_t)R * D) return;
    int r = (int)(i / D), d = (int)(i % D);
    float v = in[i];
    atomicAdd(&out[(size_t)idx[r] * D + d], sq ? v * v : v);
}
__global__ void __launch_bounds__(256)
k_seg_max(const float* in, const int* idx, float* out, int R, int D) {
    size_t i = (size_t)blockIdx.x * 256 + threadIdx.x;
    if (i >= (size_t)R * D) return;
    int r = (int)(i / D), d = (int)(i % D);
    atomMaxF(&out[(size_t)idx[r] * D + d], in[i]);
}
__global__ void __launch_bounds__(256)
k_seg_min(const float* in, const int* idx, float* out, int R, int D) {
    size_t i = (size_t)blockIdx.x * 256 + threadIdx.x;
    if (i >= (size_t)R * D) return;
    int r = (int)(i / D), d = (int)(i % D);
    atomMinF(&out[(size_t)idx[r] * D + d], in[i]);
}
__global__ void __launch_bounds__(256)
k_cat2(const float* x, const int* src, const int* dst, float* cat, int E, int H) {
    size_t i = (size_t)blockIdx.x * 256 + threadIdx.x;
    if (i >= (size_t)E * 2 * H) return;
    int e = (int)(i / (2 * H)), j = (int)(i % (2 * H));
    int node = (j < H) ? dst[e] : src[e];
    int h = (j < H) ? j : j - H;
    cat[i] = x[(size_t)node * H + h];
}
__global__ void __launch_bounds__(256)
k_pna_scaled(const float* x, const float* s, const float* mx, const float* mn,
             const float* s2, const float* deg, float* out, int N, int H) {
    size_t i = (size_t)blockIdx.x * 256 + threadIdx.x;
    if (i >= (size_t)N * H) return;
    int n = (int)(i / H), h = (int)(i % H);
    float dg = deg[n];
    float cnt = fmaxf(dg, 1.0f);
    float sv = s[i];
    float mean = sv / cnt;
    float mean2 = s2[i] / cnt;
    float var = mean2 - mean * mean;
    float sd = sqrtf(fmaxf(var, 0.0f) + 1e-5f);
    float mxv = dg > 0.0f ? mx[i] : 0.0f;
    float mnv = dg > 0.0f ? mn[i] : 0.0f;
    float agg[6] = {mean, mxv, sv, sd, var, mnv};
    float lg = logf(cnt + 1.0f);
    float amp = lg / AVG_LOG, att = AVG_LOG / lg;
    float* o = out + (size_t)n * (19 * H);
    o[h] = x[i];
#pragma unroll
    for (int k = 0; k < 6; ++k) {
        o[H + k * H + h]      = agg[k];
        o[7 * H + k * H + h]  = agg[k] * amp;
        o[13 * H + k * H + h] = agg[k] * att;
    }
}
__global__ void __launch_bounds__(256)
k_tconv_alpha(const float* q, const float* k, const float* ke,
              const int* src, const int* dst, float* alpha, int E) {
    size_t i = (size_t)blockIdx.x * 256 + threadIdx.x;
    if (i >= (size_t)E * NHh) return;
    int e = (int)(i / NHh), c = (int)(i % NHh);
    const float* qp = q + (size_t)dst[e] * Hh + c * HDh;
    const float* kp = k + (size_t)src[e] * Hh + c * HDh;
    const float* ep = ke + (size_t)e * Hh + c * HDh;
    float s = 0.f;
    for (int d = 0; d < HDh; ++d) s += qp[d] * (kp[d] + ep[d]);
    alpha[i] = s * 0.18569534f;  // 1/sqrt(29)
}
// edge softmax (R rows; rows >= E are virtual self loops i-E)
__global__ void __launch_bounds__(256)
k_sm_max(const float* a, const int* dst, float* m, int R, int E) {
    size_t i = (size_t)blockIdx.x * 256 + threadIdx.x;
    if (i >= (size_t)R * NHh) return;
    int r = (int)(i / NHh), c = (int)(i % NHh);
    int d = (r < E) ? dst[r] : (r - E);
    atomMaxF(&m[(size_t)d * NHh + c], a[i]);
}
__global__ void __launch_bounds__(256)
k_sm_exp(float* a, const int* dst, const float* m, float* z, int R, int E) {
    size_t i = (size_t)blockIdx.x * 256 + threadIdx.x;
    if (i >= (size_t)R * NHh) return;
    int r = (int)(i / NHh), c = (int)(i % NHh);
    int d = (r < E) ? dst[r] : (r - E);
    float v = expf(a[i] - m[(size_t)d * NHh + c]);
    a[i] = v;
    atomicAdd(&z[(size_t)d * NHh + c], v);
}
__global__ void __launch_bounds__(256)
k_sm_div(float* a, const int* dst, const float* z, int R, int E) {
    size_t i = (size_t)blockIdx.x * 256 + threadIdx.x;
    if (i >= (size_t)R * NHh) return;
    int r = (int)(i / NHh), c = (int)(i % NHh);
    int d = (r < E) ? dst[r] : (r - E);
    a[i] /= z[(size_t)d * NHh + c];
}
__global__ void __launch_bounds__(256)
k_tconv_agg(const float* v, const float* ke, const float* alpha,
            const int* src, const int* dst, float* out, int E) {
    size_t i = (size_t)blockIdx.x * 256 + threadIdx.x;
    if (i >= (size_t)E * Hh) return;
    int e = (int)(i / Hh), h = (int)(i % Hh);
    int c = h / HDh;
    float t = (v[(size_t)src[e] * Hh + h] + ke[i]) * alpha[(size_t)e * NHh + c];
    atomicAdd(&out[(size_t)dst[e] * Hh + h], t);
}
__global__ void __launch_bounds__(256)
k_dot_heads(const float* X, const float* att, float* out, int R) {
    size_t i = (size_t)blockIdx.x * 256 + threadIdx.x;
    if (i >= (size_t)R * NHh) return;
    int r = (int)(i / NHh), c = (int)(i % NHh);
    const float* xp = X + (size_t)r * Hh + c * HDh;
    const float* ap = att + c * HDh;
    float s = 0.f;
    for (int d = 0; d < HDh; ++d) s += xp[d] * ap[d];
    out[i] = s;
}
__global__ void __launch_bounds__(256)
k_div_deg(float* Y, const float* deg, int N, int D) {
    size_t i = (size_t)blockIdx.x * 256 + threadIdx.x;
    if (i >= (size_t)N * D) return;
    Y[i] /= fmaxf(deg[i / D], 1.0f);
}
__global__ void __launch_bounds__(256)
k_concat_rows(const float* ea, const float* la, float* e2, int E, int N, int H) {
    size_t i = (size_t)blockIdx.x * 256 + threadIdx.x;
    if (i >= (size_t)(E + N) * H) return;
    int r = (int)(i / H), h = (int)(i % H);
    e2[i] = (r < E) ? ea[i] : la[(size_t)(r - E) * H + h];
}
__global__ void __launch_bounds__(256)
k_gat_logits(const float* asrc, const float* adst, const float* aed,
             const int* src, const int* dst, float* alpha, int R, int E) {
    size_t i = (size_t)blockIdx.x * 256 + threadIdx.x;
    if (i >= (size_t)R * NHh) return;
    int r = (int)(i / NHh), c = (int)(i % NHh);
    int s = (r < E) ? src[r] : (r - E);
    int d = (r < E) ? dst[r] : (r - E);
    float v = asrc[(size_t)s * NHh + c] + adst[(size_t)d * NHh + c] + aed[i];
    alpha[i] = v > 0.f ? v : 0.2f * v;
}
__global__ void __launch_bounds__(256)
k_gat_agg(const float* xh, const float* alpha, const int* src, const int* dst,
          float* out, int R, int E) {
    size_t i = (size_t)blockIdx.x * 256 + threadIdx.x;
    if (i >= (size_t)R * Hh) return;
    int r = (int)(i / Hh), h = (int)(i % Hh);
    int s = (r < E) ? src[r] : (r - E);
    int d = (r < E) ? dst[r] : (r - E);
    int c = h / HDh;
    atomicAdd(&out[(size_t)d * Hh + h],
              xh[(size_t)s * Hh + h] * alpha[(size_t)r * NHh + c]);
}
__global__ void __launch_bounds__(256)
k_gine_msg(const float* x, const float* ea, const int* src, const int* dst,
           float* agg, int E, int H) {
    size_t i = (size_t)blockIdx.x * 256 + threadIdx.x;
    if (i >= (size_t)E * H) return;
    int e = (int)(i / H), h = (int)(i % H);
    float v = fmaxf(x[(size_t)src[e] * H + h] + ea[i], 0.0f);
    atomicAdd(&agg[(size_t)dst[e] * H + h], v);
}
__global__ void __launch_bounds__(256)
k_gcn_dinv(const float* deg, float* dinv, int N) {
    int n = blockIdx.x * 256 + threadIdx.x;
    if (n < N) dinv[n] = rsqrtf(deg[n] + 1.0f);
}
__global__ void __launch_bounds__(256)
k_gcn_edge(const float* xw, const float* dinv, const int* src, const int* dst,
           float* out, int E, int H) {
    size_t i = (size_t)blockIdx.x * 256 + threadIdx.x;
    if (i >= (size_t)E * H) return;
    int e = (int)(i / H), h = (int)(i % H);
    int s = src[e], d = dst[e];
    atomicAdd(&out[(size_t)d * H + h], xw[(size_t)s * H + h] * dinv[s] * dinv[d]);
}
__global__ void __launch_bounds__(256)
k_gcn_self(const float* xw, const float* dinv, float* out, int N, int H) {
    size_t i = (size_t)blockIdx.x * 256 + threadIdx.x;
    if (i >= (size_t)N * H) return;
    int n = (int)(i / H);
    out[i] += xw[i] * dinv[n] * dinv[n];
}
__global__ void __launch_bounds__(256)
k_cnt(const int* batch, float* cnt, int N) {
    int n = blockIdx.x * 256 + threadIdx.x;
    if (n < N) atomicAdd(&cnt[batch[n]], 1.0f);
}
__global__ void __launch_bounds__(256)
k_mean(const float* sm, const float* cnt, float* mean, int G, int H) {
    size_t i = (size_t)blockIdx.x * 256 + threadIdx.x;
    if (i >= (size_t)G * H) return;
    mean[i] = sm[i] / fmaxf(cnt[i / H], 1.0f);
}
__global__ void __launch_bounds__(256)
k_std_acc(const float* x, const float* mean, const int* batch, float* ss,
          int N, int H) {
    size_t i = (size_t)blockIdx.x * 256 + threadIdx.x;
    if (i >= (size_t)N * H) return;
    int n = (int)(i / H), h = (int)(i % H);
    float d = x[i] - mean[(size_t)batch[n] * H + h];
    atomicAdd(&ss[(size_t)batch[n] * H + h], d * d);
}
__global__ void __launch_bounds__(256)
k_pool_fin(const float* ss, const float* cnt, float* mx, float* sd, int G, int H) {
    size_t i = (size_t)blockIdx.x * 256 + threadIdx.x;
    if (i >= (size_t)G * H) return;
    float c = cnt[i / H];
    sd[i] = sqrtf(ss[i] / fmaxf(c, 1.0f));
    if (!(c > 0.0f)) mx[i] = 0.0f;
}

// ---------------------------------------------------------------------------
// Host orchestration
// ---------------------------------------------------------------------------
extern "C" void kernel_launch(void* const* d_in, const int* in_sizes, int n_in,
                              void* d_out, int out_size, void* d_ws, size_t ws_size,
                              hipStream_t stream) {
    (void)in_sizes; (void)n_in; (void)out_size; (void)ws_size;
    const float* x6    = (const float*)d_in[0];
    const float* eattr = (const float*)d_in[1];
    const int*   eidx  = (const int*)d_in[2];
    const int*   batch = (const int*)d_in[3];
    const int*   srcI  = eidx;
    const int*   dstI  = eidx + En;

    // ---- flattened params (jax pytree: dict keys alphabetical) ----
    int pi = 4;
    auto nP = [&]() { return (const float*)d_in[pi++]; };
    // adapter: lin.b, lin.w, ln.b, ln.g
    const float *ad_lin_b = nP(), *ad_lin_w = nP(), *ad_ln_b = nP(), *ad_ln_g = nP();
    // atom: l1.b,l1.w,l2.b,l2.w,l3.b,l3.w,n1.b,n1.g,n2.b,n2.g
    const float *at_l1_b = nP(), *at_l1_w = nP(), *at_l2_b = nP(), *at_l2_w = nP();
    const float *at_l3_b = nP(), *at_l3_w = nP(), *at_n1_b = nP(), *at_n1_g = nP();
    const float *at_n2_b = nP(), *at_n2_g = nP();
    // attn: o.b,o.w,v.b,v.w
    const float *an_o_b = nP(), *an_o_w = nP(), *an_v_b = nP(), *an_v_w = nP();
    // edge: l1.b,l1.w,l2.b,l2.w,n1.b,n1.g,n2.b,n2.g
    const float *ed_l1_b = nP(), *ed_l1_w = nP(), *ed_l2_b = nP(), *ed_l2_w = nP();
    const float *ed_n1_b = nP(), *ed_n1_g = nP(), *ed_n2_b = nP(), *ed_n2_g = nP();
    // layers
    struct Lyr {
        const float *bn_b, *bn_g;
        const float *c[10];
        const float *ff_l1_b, *ff_l1_w, *ff_l2_b, *ff_l2_w;
        const float *ff_n1_b, *ff_n1_g, *ff_n2_b, *ff_n2_g;
        const float *res_b, *res_w;
    };
    static const int convLeaves[5] = {6, 9, 6, 8, 2};
    Lyr L[12];
    for (int i = 0; i < 12; ++i) {
        int t = i % 5;
        L[i].bn_b = nP(); L[i].bn_g = nP();
        for (int j = 0; j < convLeaves[t]; ++j) L[i].c[j] = nP();
        L[i].ff_l1_b = nP(); L[i].ff_l1_w = nP();
        L[i].ff_l2_b = nP(); L[i].ff_l2_w = nP();
        L[i].ff_n1_b = nP(); L[i].ff_n1_g = nP();
        L[i].ff_n2_b = nP(); L[i].ff_n2_g = nP();
        L[i].res_b = nP(); L[i].res_w = nP();
    }
    // pool (alphabetical: add, max, mean, std), each l.b,l.w,n.b,n.g
    const float *pl_b[4], *pl_w[4], *pn_b[4], *pn_g[4];
    for (int j = 0; j < 4; ++j) { pl_b[j] = nP(); pl_w[j] = nP(); pn_b[j] = nP(); pn_g[j] = nP(); }

    // ---- workspace bump allocator (deterministic each call) ----
    char* wp = (char*)d_ws;
    auto alloc = [&](size_t bytes) -> char* {
        char* p = wp; wp += (bytes + 255) & ~(size_t)255; return p;
    };
    auto allocF = [&](size_t n) { return (float*)alloc(n * 4); };

    auto g1 = [&](size_t n) { return (unsigned)((n + 255) / 256); };
    auto zero = [&](float* p, size_t n) { hipMemsetAsync(p, 0, n * 4, stream); };
    const float FINF = __builtin_inff();

    auto gemm = [&](const float* A, int M, int K, const float* W, int Ncols,
                    const float* bias, float* C, int act) {
        int Kp = ((K + 63) / 64) * 64;          // pad K to BK for async B staging
        int Nt = ((Ncols + 127) / 128) * 128;
        __bf16* Wt = (__bf16*)alloc((size_t)Nt * Kp * 2);
        size_t tw = (size_t)Nt * Kp;
        k_prep_w<<<g1(tw), 256, 0, stream>>>(W, Wt, K, Ncols, Kp, Nt);
        dim3 grid((Ncols + BN - 1) / BN, (M + BM - 1) / BM);
        k_gemm<<<grid, 256, 0, stream>>>(A, Wt, bias, C, M, Ncols, K, Kp, act);
    };
    auto lnrm = [&](const float* X, int rows, int D, const float* g_, const float* b_,
                    const float* add, float* Y, int post, int out_ld, int out_off) {
        k_ln<<<rows, 256, 0, stream>>>(X, g_, b_, add, Y, D, out_ld, out_off, post);
    };

    // ---- activation buffers ----
    float* x12    = allocF((size_t)Nn * 12);
    float* t1     = allocF((size_t)Nn * FFh);
    float* hb     = allocF((size_t)Nn * Hh);
    float* xa     = allocF((size_t)Nn * Hh);
    float* xb     = allocF((size_t)Nn * Hh);
    float* resb   = allocF((size_t)Nn * Hh);
    float* t2     = allocF((size_t)Nn * Hh);
    float* eat    = allocF((size_t)En * 174);
    float* ea     = allocF((size_t)En * Hh);
    float* deg    = allocF(Nn);
    float* dinv   = allocF(Nn);
    float* cat2   = allocF((size_t)En * 2 * Hh);
    float* hE     = allocF((size_t)En * Hh);
    float* segS   = allocF((size_t)Nn * Hh);
    float* segMx  = allocF((size_t)Nn * Hh);
    float* segMn  = allocF((size_t)Nn * Hh);
    float* segS2  = allocF((size_t)Nn * Hh);
    float* scaled = allocF((size_t)Nn * 19 * Hh);
    float* qb     = allocF((size_t)Nn * Hh);
    float* kb     = allocF((size_t)Nn * Hh);
    float* vb     = allocF((size_t)Nn * Hh);
    float* alb    = allocF((size_t)(En + Nn) * NHh);
    float* msg    = allocF((size_t)Nn * NHh);
    float* zsg    = allocF((size_t)Nn * NHh);
    float* e2     = allocF((size_t)(En + Nn) * Hh);
    float* eh     = allocF((size_t)(En + Nn) * Hh);
    float* aed    = allocF((size_t)(En + Nn) * NHh);
    float* asb    = allocF((size_t)Nn * NHh);
    float* adb    = allocF((size_t)Nn * NHh);
    float* cntG   = allocF(Gn);
    float* smG    = allocF((size_t)Gn * Hh);
    float* mxG    = allocF((size_t)Gn * Hh);
    float* ssG    = allocF((size_t)Gn * Hh);
    float* meanG  = allocF((size_t)Gn * Hh);
    float* stdG   = allocF((size_t)Gn * Hh);
    float* ptG    = allocF((size_t)Gn * Hh);
    float* feats  = allocF((size_t)Gn * 4 * Hh);
    float* av     = allocF((size_t)Gn * 4 * Hh);
    float* ao     = allocF((size_t)Gn * 4 * Hh);

    // ======================= encoders =======================
    gemm(x6, Nn, 6, ad_lin_w, 12, ad_lin_b, x12, 0);
    lnrm(x12, Nn, 12, ad_ln_g, ad_ln_b, nullptr, x12, 1, 12, 0);
    gemm(x12, Nn, 12, at_l1_w, FFh, at_l1_b, t1, 0);
    lnrm(t1, Nn, FFh, at_n1_g, at_n1_b, nullptr, t1, 1, FFh, 0);
    gemm(t1, Nn, FFh, at_l2_w, Hh, at_l2_b, hb, 0);
    lnrm(hb, Nn, Hh, at_n2_g, at_n2_b, nullptr, hb, 1, Hh, 0);
    gemm(hb, Nn, Hh, at_l3_w, Hh, at_l3_b, xa, 0);

    gemm(eattr, En, 6, ed_l1_w, 174, ed_l1_b, eat, 0);
    lnrm(eat, En, 174, ed_n1_g, ed_n1_b, nullptr, eat, 1, 174, 0);
    gemm(eat, En, 174, ed_l2_w, Hh, ed_l2_b, ea, 0);
    lnrm(ea, En, Hh, ed_n2_g, ed_n2_b, nullptr, ea, 0, Hh, 0);

    zero(deg, Nn);
    k_deg<<<g1(En), 256, 0, stream>>>(dstI, deg, En);

    // ======================= layers =========================
    float* xc = xa;
    float* xo = xb;
    for (int i = 0; i < 12; ++i) {
        int t = i % 5;
        Lyr& l = L[i];
        if (t == 0) {  // PNA: lin.b,lin.w,post.b,post.w,pre.b,pre.w
            k_cat2<<<g1((size_t)En * 2 * Hh), 256, 0, stream>>>(xc, srcI, dstI, cat2, En, Hh);
            gemm(cat2, En, 2 * Hh, l.c[5], Hh, l.c[4], hE, 0);
            zero(segS, (size_t)Nn * Hh); zero(segS2, (size_t)Nn * Hh);
            k_fill<<<g1((size_t)Nn * Hh), 256, 0, stream>>>(segMx, -FINF, (size_t)Nn * Hh);
            k_fill<<<g1((size_t)Nn * Hh), 256, 0, stream>>>(segMn,  FINF, (size_t)Nn * Hh);
            k_seg_sum<<<g1((size_t)En * Hh), 256, 0, stream>>>(hE, dstI, segS, En, Hh, 0);
            k_seg_sum<<<g1((size_t)En * Hh), 256, 0, stream>>>(hE, dstI, segS2, En, Hh, 1);
            k_seg_max<<<g1((size_t)En * Hh), 256, 0, stream>>>(hE, dstI, segMx, En, Hh);
            k_seg_min<<<g1((size_t)En * Hh), 256, 0, stream>>>(hE, dstI, segMn, En, Hh);
            k_pna_scaled<<<g1((size_t)Nn * Hh), 256, 0, stream>>>(xc, segS, segMx, segMn, segS2, deg, scaled, Nn, Hh);
            gemm(scaled, Nn, 19 * Hh, l.c[3], Hh, l.c[2], t2, 0);
            gemm(t2, Nn, Hh, l.c[1], Hh, l.c[0], hb, 0);
        } else if (t == 1) {  // TransformerConv: e.w,k.b,k.w,q.b,q.w,skip.b,skip.w,v.b,v.w
            gemm(xc, Nn, Hh, l.c[4], Hh, l.c[3], qb, 0);
            gemm(xc, Nn, Hh, l.c[2], Hh, l.c[1], kb, 0);
            gemm(xc, Nn, Hh, l.c[8], Hh, l.c[7], vb, 0);
            gemm(ea, En, Hh, l.c[0], Hh, nullptr, hE, 0);  // ke
            k_tconv_alpha<<<g1((size_t)En * NHh), 256, 0, stream>>>(qb, kb, hE, srcI, dstI, alb, En);
            k_fill<<<g1((size_t)Nn * NHh), 256, 0, stream>>>(msg, -FINF, (size_t)Nn * NHh);
            zero(zsg, (size_t)Nn * NHh);
            k_sm_max<<<g1((size_t)En * NHh), 256, 0, stream>>>(alb, dstI, msg, En, En);
            k_sm_exp<<<g1((size_t)En * NHh), 256, 0, stream>>>(alb, dstI, msg, zsg, En, En);
            k_sm_div<<<g1((size_t)En * NHh), 256, 0, stream>>>(alb, dstI, zsg, En, En);
            zero(hb, (size_t)Nn * Hh);
            k_tconv_agg<<<g1((size_t)En * Hh), 256, 0, stream>>>(vb, hE, alb, srcI, dstI, hb, En);
            gemm(xc, Nn, Hh, l.c[6], Hh, l.c[5], t2, 0);   // skip
            k_add<<<g1((size_t)Nn * Hh), 256, 0, stream>>>(hb, t2, hb, (size_t)Nn * Hh);
        } else if (t == 2) {  // GAT: att_dst,att_edge,att_src,bias,e.w,lin.w
            gemm(xc, Nn, Hh, l.c[5], Hh, nullptr, qb, 0);  // xh
            k_dot_heads<<<g1((size_t)Nn * NHh), 256, 0, stream>>>(qb, l.c[2], asb, Nn);
            k_dot_heads<<<g1((size_t)Nn * NHh), 256, 0, stream>>>(qb, l.c[0], adb, Nn);
            zero(segS, (size_t)Nn * Hh);
            k_seg_sum<<<g1((size_t)En * Hh), 256, 0, stream>>>(ea, dstI, segS, En, Hh, 0);
            k_div_deg<<<g1((size_t)Nn * Hh), 256, 0, stream>>>(segS, deg, Nn, Hh);  // loop_attr
            k_concat_rows<<<g1((size_t)(En + Nn) * Hh), 256, 0, stream>>>(ea, segS, e2, En, Nn, Hh);
            gemm(e2, En + Nn, Hh, l.c[4], Hh, nullptr, eh, 0);
            k_dot_heads<<<g1((size_t)(En + Nn) * NHh), 256, 0, stream>>>(eh, l.c[1], aed, En + Nn);
            k_gat_logits<<<g1((size_t)(En + Nn) * NHh), 256, 0, stream>>>(asb, adb, aed, srcI, dstI, alb, En + Nn, En);
            k_fill<<<g1((size_t)Nn * NHh), 256, 0, stream>>>(msg, -FINF, (size_t)Nn * NHh);
            zero(zsg, (size_t)Nn * NHh);
            k_sm_max<<<g1((size_t)(En + Nn) * NHh), 256, 0, stream>>>(alb, dstI, msg, En + Nn, En);
            k_sm_exp<<<g1((size_t)(En + Nn) * NHh), 256, 0, stream>>>(alb, dstI, msg, zsg, En + Nn, En);
            k_sm_div<<<g1((size_t)(En + Nn) * NHh), 256, 0, stream>>>(alb, dstI, zsg, En + Nn, En);
            zero(hb, (size_t)Nn * Hh);
            k_gat_agg<<<g1((size_t)(En + Nn) * Hh), 256, 0, stream>>>(qb, alb, srcI, dstI, hb, En + Nn, En);
            k_add_rowvec<<<g1((size_t)Nn * Hh), 256, 0, stream>>>(hb, l.c[3], Nn, Hh);
        } else if (t == 3) {  // GINE: l1.b,l1.w,l2.b,l2.w,n1.b,n1.g,n2.b,n2.g
            zero(segS, (size_t)Nn * Hh);
            k_gine_msg<<<g1((size_t)En * Hh), 256, 0, stream>>>(xc, ea, srcI, dstI, segS, En, Hh);
            k_add<<<g1((size_t)Nn * Hh), 256, 0, stream>>>(xc, segS, t2, (size_t)Nn * Hh);
            gemm(t2, Nn, Hh, l.c[1], FFh, l.c[0], t1, 0);
            lnrm(t1, Nn, FFh, l.c[5], l.c[4], nullptr, t1, 1, FFh, 0);
            gemm(t1, Nn, FFh, l.c[3], Hh, l.c[2], hb, 0);
            lnrm(hb, Nn, Hh, l.c[7], l.c[6], nullptr, hb, 0, Hh, 0);
        } else {  // GCN: b, w
            gemm(xc, Nn, Hh, l.c[1], Hh, nullptr, qb, 0);  // xw
            k_gcn_dinv<<<g1(Nn), 256, 0, stream>>>(deg, dinv, Nn);
            zero(hb, (size_t)Nn * Hh);
            k_gcn_edge<<<g1((size_t)En * Hh), 256, 0, stream>>>(qb, dinv, srcI, dstI, hb, En, Hh);
            k_gcn_self<<<g1((size_t)Nn * Hh), 256, 0, stream>>>(qb, dinv, hb, Nn, Hh);
            k_add_rowvec<<<g1((size_t)Nn * Hh), 256, 0, stream>>>(hb, l.c[0], Nn, Hh);
        }
        // combine + FF
        gemm(xc, Nn, Hh, l.res_w, Hh, l.res_b, resb, 0);
        lnrm(hb, Nn, Hh, l.bn_g, l.bn_b, resb, xo, 0, Hh, 0);
        gemm(xo, Nn, Hh, l.ff_l1_w, FFh, l.ff_l1_b, t1, 0);
        lnrm(t1, Nn, FFh, l.ff_n1_g, l.ff_n1_b, nullptr, t1, 1, FFh, 0);
        gemm(t1, Nn, FFh, l.ff_l2_w, Hh, l.ff_l2_b, hb, 0);
        lnrm(hb, Nn, Hh, l.ff_n2_g, l.ff_n2_b, xo, xo, 1, Hh, 0);
        float* tmp = xc; xc = xo; xo = tmp;
    }

    // ======================= pooling ========================
    zero(cntG, Gn); zero(smG, (size_t)Gn * Hh); zero(ssG, (size_t)Gn * Hh);
    k_fill<<<g1((size_t)Gn * Hh), 256, 0, stream>>>(mxG, -FINF, (size_t)Gn * Hh);
    k_cnt<<<g1(Nn), 256, 0, stream>>>(batch, cntG, Nn);
    k_seg_sum<<<g1((size_t)Nn * Hh), 256, 0, stream>>>(xc, batch, smG, Nn, Hh, 0);
    k_seg_max<<<g1((size_t)Nn * Hh), 256, 0, stream>>>(xc, batch, mxG, Nn, Hh);
    k_mean<<<g1((size_t)Gn * Hh), 256, 0, stream>>>(smG, cntG, meanG, Gn, Hh);
    k_std_acc<<<g1((size_t)Nn * Hh), 256, 0, stream>>>(xc, meanG, batch, ssG, Nn, Hh);
    k_pool_fin<<<g1((size_t)Gn * Hh), 256, 0, stream>>>(ssG, cntG, mxG, stdG, Gn, Hh);

    // feats = [gelu(ln(lin(mean))), ..max.., ..add.., ..std..]
    const float* pin[4] = {meanG, mxG, smG, stdG};
    const int pmap[4] = {2, 1, 0, 3};  // params sorted add,max,mean,std
    for (int j = 0; j < 4; ++j) {
        int pj = pmap[j];
        gemm(pin[j], Gn, Hh, pl_w[pj], Hh, pl_b[pj], ptG, 0);
        lnrm(ptG, Gn, Hh, pn_g[pj], pn_b[pj], nullptr, feats, 1, 4 * Hh, j * Hh);
    }
    // attn over seq_len 1: out = feats + o(v(feats))
    gemm(feats, Gn, 4 * Hh, an_v_w, 4 * Hh, an_v_b, av, 0);
    gemm(av, Gn, 4 * Hh, an_o_w, 4 * Hh, an_o_b, ao, 0);
    k_add<<<g1((size_t)Gn * 4 * Hh), 256, 0, stream>>>(feats, ao, (float*)d_out, (size_t)Gn * 4 * Hh);
}
